// HDMOEM_10840497455465
// MI455X (gfx1250) — matrix operand order repaired and checked by reading, not compiled, not run on hardware
//
#include <hip/hip_runtime.h>
#include <hip/hip_fp16.h>

typedef __attribute__((ext_vector_type(16))) _Float16 v16h;
typedef __attribute__((ext_vector_type(8)))  float    v8f;

union Frag16 { v16h v; unsigned int u[8]; };

__device__ __forceinline__ float mp_silu_f(float x) {
    return x * (1.0f / (1.0f + __expf(-x))) * (1.0f / 0.596f);
}

// ---------------------------------------------------------------------------
// Weight normalization (mpw): each row divided by sqrt(sum(w^2) + 1e-8*fanin)
// ---------------------------------------------------------------------------
__global__ void norm_rows_f32(const float* __restrict__ w, float* __restrict__ o, int R, int F) {
    __shared__ float red[256];
    int row = blockIdx.x;
    const float* p = w + (long)row * F;
    float ss = 0.f;
    for (int i = threadIdx.x; i < F; i += 256) { float v = p[i]; ss += v * v; }
    red[threadIdx.x] = ss; __syncthreads();
    for (int k = 128; k > 0; k >>= 1) { if (threadIdx.x < k) red[threadIdx.x] += red[threadIdx.x + k]; __syncthreads(); }
    float inv = rsqrtf(red[0] + 1e-8f * (float)F);
    for (int i = threadIdx.x; i < F; i += 256) o[(long)row * F + i] = p[i] * inv;
}

__global__ void norm_rows_f16(const float* __restrict__ w, _Float16* __restrict__ o, int R, int F) {
    __shared__ float red[256];
    int row = blockIdx.x;
    const float* p = w + (long)row * F;
    float ss = 0.f;
    for (int i = threadIdx.x; i < F; i += 256) { float v = p[i]; ss += v * v; }
    red[threadIdx.x] = ss; __syncthreads();
    for (int k = 128; k > 0; k >>= 1) { if (threadIdx.x < k) red[threadIdx.x] += red[threadIdx.x + k]; __syncthreads(); }
    float inv = rsqrtf(red[0] + 1e-8f * (float)F);
    for (int i = threadIdx.x; i < F; i += 256) o[(long)row * F + i] = (_Float16)(p[i] * inv);
}

// uw_mid [e][n][c][3][3] -> normalized f16 [e*192+n][k] with k = tap*192 + c
__global__ void norm_repack_uwmid(const float* __restrict__ w, _Float16* __restrict__ o) {
    __shared__ float red[256];
    int row = blockIdx.x;                      // e*192 + n
    const float* p = w + (long)row * 1728;
    float ss = 0.f;
    for (int i = threadIdx.x; i < 1728; i += 256) { float v = p[i]; ss += v * v; }
    red[threadIdx.x] = ss; __syncthreads();
    for (int k = 128; k > 0; k >>= 1) { if (threadIdx.x < k) red[threadIdx.x] += red[threadIdx.x + k]; __syncthreads(); }
    float inv = rsqrtf(red[0] + 1e-8f * 1728.f);
    for (int i = threadIdx.x; i < 1728; i += 256) {
        int c = i / 9, tap = i - 9 * c;
        o[(long)row * 1728 + tap * 192 + c] = (_Float16)(p[i] * inv);
    }
}

// ---------------------------------------------------------------------------
// Generic WMMA GEMM: C[M,N](f16) = epi(A[M,K](f16) @ W[N,K]^T(f16))
// block = 128 threads (4 waves), tile 64x64, K step 32. Batched via grid.z.
// ---------------------------------------------------------------------------
#define LDSW 56

template<int EPI>
__global__ __launch_bounds__(128) void gemm_wmma_kernel(
    const _Float16* __restrict__ A, long sA,
    const _Float16* __restrict__ W, long sW,
    _Float16* __restrict__ C, long sC,
    int M, int N, int K)
{
    __shared__ __align__(16) _Float16 lA[64 * LDSW];
    __shared__ __align__(16) _Float16 lB[64 * LDSW];
    const int e = blockIdx.z;
    A += (long)e * sA; W += (long)e * sW; C += (long)e * sC;
    const int m0 = blockIdx.x * 64, n0 = blockIdx.y * 64;
    const int tid  = threadIdx.x;
    const int wave = tid >> 5, lane = tid & 31;
    const int half = lane >> 4, r = lane & 15;
    const int lrow = tid >> 1, lpart = (tid & 1) * 16;
    v8f acc0 = {}, acc1 = {}, acc2 = {}, acc3 = {};

    for (int kk = 0; kk < K; kk += 32) {
        const _Float16* pa = A + (long)(m0 + lrow) * K + kk + lpart;
        const _Float16* pb = W + (long)(n0 + lrow) * K + kk + lpart;
        if (kk + 32 < K) { __builtin_prefetch((const void*)(pa + 32), 0, 0); __builtin_prefetch((const void*)(pb + 32), 0, 0); }
        *(uint4*)&lA[lrow * LDSW + lpart]     = *(const uint4*)pa;
        *(uint4*)&lA[lrow * LDSW + lpart + 8] = *(const uint4*)(pa + 8);
        *(uint4*)&lB[lrow * LDSW + lpart]     = *(const uint4*)pb;
        *(uint4*)&lB[lrow * LDSW + lpart + 8] = *(const uint4*)(pb + 8);
        __syncthreads();
        Frag16 fa;
        #pragma unroll
        for (int v = 0; v < 8; ++v) {
            int kv = (v < 4 ? 2 * v : 8 + 2 * v) + 8 * half;   // ISA 16-bit A 16x32 layout
            fa.u[v] = *(const unsigned int*)&lA[(wave * 16 + r) * LDSW + kv];
        }
        #pragma unroll
        for (int sub = 0; sub < 4; ++sub) {
            Frag16 fb;
            #pragma unroll
            for (int v = 0; v < 8; ++v) {
                int kv = (v < 4 ? 2 * v : 8 + 2 * v) + 8 * half; // same k order as A fragment
                fb.u[v] = *(const unsigned int*)&lB[(sub * 16 + r) * LDSW + kv];
            }
            v8f& a = (sub == 0 ? acc0 : sub == 1 ? acc1 : sub == 2 ? acc2 : acc3);
            a = __builtin_amdgcn_wmma_f32_16x16x32_f16(false, fa.v, false, fb.v, (short)0, a, false, false);
        }
        __syncthreads();
    }
    #pragma unroll
    for (int sub = 0; sub < 4; ++sub) {
        v8f a = (sub == 0 ? acc0 : sub == 1 ? acc1 : sub == 2 ? acc2 : acc3);
        #pragma unroll
        for (int rr = 0; rr < 8; ++rr) {
            int row = m0 + wave * 16 + rr + 8 * half;          // ISA f32 C/D layout
            int col = n0 + sub * 16 + r;
            float vv = a[rr];
            if (EPI == 1) vv = mp_silu_f(vv);
            C[(long)row * N + col] = (_Float16)vv;
        }
    }
}

// ---------------------------------------------------------------------------
// UNet mid conv as implicit GEMM: h2[16384,192] = mp_silu(im2col(h1) @ W^T)
// h1: [b][y][x][192] f16 (per expert), W: [192][1728] f16, K = tap*192 + c
// ---------------------------------------------------------------------------
__global__ __launch_bounds__(128) void conv_mid_wmma_kernel(
    const _Float16* __restrict__ h1,
    const _Float16* __restrict__ W,
    _Float16* __restrict__ h2)
{
    __shared__ __align__(16) _Float16 lA[64 * LDSW];
    __shared__ __align__(16) _Float16 lB[64 * LDSW];
    const int m0 = blockIdx.x * 64, n0 = blockIdx.y * 64;
    const int tid  = threadIdx.x;
    const int wave = tid >> 5, lane = tid & 31;
    const int half = lane >> 4, r = lane & 15;
    const int lrow = tid >> 1, lpart = (tid & 1) * 16;
    const int m = m0 + lrow;
    const int b = m >> 12, rem = m & 4095, y = rem >> 6, x = rem & 63;
    v8f acc0 = {}, acc1 = {}, acc2 = {}, acc3 = {};

    for (int kk = 0; kk < 1728; kk += 32) {
        int tap = kk / 192, c0 = kk - tap * 192;
        int ty = y + tap / 3 - 1, tx = x + tap % 3 - 1;
        if (ty >= 0 && ty < 64 && tx >= 0 && tx < 64) {
            const _Float16* pa = h1 + (long)((b * 64 + ty) * 64 + tx) * 192 + c0 + lpart;
            *(uint4*)&lA[lrow * LDSW + lpart]     = *(const uint4*)pa;
            *(uint4*)&lA[lrow * LDSW + lpart + 8] = *(const uint4*)(pa + 8);
        } else {
            uint4 z; z.x = 0u; z.y = 0u; z.z = 0u; z.w = 0u;
            *(uint4*)&lA[lrow * LDSW + lpart]     = z;
            *(uint4*)&lA[lrow * LDSW + lpart + 8] = z;
        }
        const _Float16* pb = W + (long)(n0 + lrow) * 1728 + kk + lpart;
        *(uint4*)&lB[lrow * LDSW + lpart]     = *(const uint4*)pb;
        *(uint4*)&lB[lrow * LDSW + lpart + 8] = *(const uint4*)(pb + 8);
        __syncthreads();
        Frag16 fa;
        #pragma unroll
        for (int v = 0; v < 8; ++v) {
            int kv = (v < 4 ? 2 * v : 8 + 2 * v) + 8 * half;
            fa.u[v] = *(const unsigned int*)&lA[(wave * 16 + r) * LDSW + kv];
        }
        #pragma unroll
        for (int sub = 0; sub < 4; ++sub) {
            Frag16 fb;
            #pragma unroll
            for (int v = 0; v < 8; ++v) {
                int kv = (v < 4 ? 2 * v : 8 + 2 * v) + 8 * half;
                fb.u[v] = *(const unsigned int*)&lB[(sub * 16 + r) * LDSW + kv];
            }
            v8f& a = (sub == 0 ? acc0 : sub == 1 ? acc1 : sub == 2 ? acc2 : acc3);
            a = __builtin_amdgcn_wmma_f32_16x16x32_f16(false, fa.v, false, fb.v, (short)0, a, false, false);
        }
        __syncthreads();
    }
    #pragma unroll
    for (int sub = 0; sub < 4; ++sub) {
        v8f a = (sub == 0 ? acc0 : sub == 1 ? acc1 : sub == 2 ? acc2 : acc3);
        #pragma unroll
        for (int rr = 0; rr < 8; ++rr) {
            int row = m0 + wave * 16 + rr + 8 * half;
            int col = n0 + sub * 16 + r;
            h2[(long)row * 192 + col] = (_Float16)mp_silu_f(a[rr]);
        }
    }
}

// ---------------------------------------------------------------------------
// Small VALU kernels
// ---------------------------------------------------------------------------
__global__ void femb_kernel(const float* __restrict__ tv, const float* __restrict__ fr,
                            const float* __restrict__ ph, float* __restrict__ femb) {
    int idx = blockIdx.x * 256 + threadIdx.x;
    if (idx >= 1024) return;
    int b = idx >> 8, i = idx & 255;
    femb[idx] = 1.41421356237f * cosf(6.283185307179586f * (tv[b] * fr[i] + ph[i]));
}

__global__ void small_linear_kernel(const float* __restrict__ in, const float* __restrict__ W,
                                    float* __restrict__ out, int Bn, int O, int K, int act) {
    int idx = blockIdx.x * 256 + threadIdx.x;
    if (idx >= Bn * O) return;
    int b = idx / O, o = idx - b * O;
    const float* ip = in + (long)b * K;
    const float* wp = W + (long)o * K;
    float acc = 0.f;
    for (int k = 0; k < K; ++k) acc += ip[k] * wp[k];
    if (act == 1) acc = mp_silu_f(acc);
    out[idx] = acc;
}

__global__ void scale_kernel(const float* __restrict__ temb, const float* __restrict__ wsc,
                             const float* __restrict__ zeta, float* __restrict__ scale) {
    __shared__ float lg[8];
    int t = threadIdx.x;
    if (t < 8) {
        int b = t >> 1, j = t & 1;
        float acc = 0.f;
        for (int k = 0; k < 512; ++k) acc += temb[b * 512 + k] * wsc[j * 512 + k];
        lg[t] = acc / zeta[0];
    }
    __syncthreads();
    if (t < 4) {
        int b = t;
        float a = lg[2 * b], c = lg[2 * b + 1];
        float m = fmaxf(a, c);
        float e0 = __expf(a - m), e1 = __expf(c - m), s = e0 + e1;
        scale[2 * b]     = 2.f * e0 / s;
        scale[2 * b + 1] = 2.f * e1 / s;
    }
}

__global__ void xmean_kernel(const float* __restrict__ x, float* __restrict__ xm) {
    __shared__ float red[256];
    int bc = blockIdx.x;
    const float* p = x + (long)bc * 4096;
    float s = 0.f;
    for (int i = threadIdx.x; i < 4096; i += 256) s += p[i];
    red[threadIdx.x] = s; __syncthreads();
    for (int k = 128; k > 0; k >>= 1) { if (threadIdx.x < k) red[threadIdx.x] += red[threadIdx.x + k]; __syncthreads(); }
    if (threadIdx.x == 0) xm[bc] = red[0] * (1.f / 4096.f);
}

__global__ void inscale_kernel(const float* __restrict__ x, const float* __restrict__ sc,
                               float* __restrict__ iv, float* __restrict__ iu) {
    int idx = blockIdx.x * 256 + threadIdx.x;
    if (idx >= 65536) return;
    int b = idx >> 14;
    float v = x[idx];
    iv[idx] = sc[b * 2 + 0] * v;   // vit gets scale[:,0]
    iu[idx] = sc[b * 2 + 1] * v;   // unet gets scale[:,1]
}

__global__ void router_kernel(const float* __restrict__ xmean, const float* __restrict__ scale, int sidx,
                              const float* __restrict__ temb, const float* __restrict__ wr,
                              const unsigned char* __restrict__ mask, const float* __restrict__ zeta,
                              float* __restrict__ rw, float* __restrict__ probs) {
    __shared__ float lg[16];
    int t = threadIdx.x;
    if (t < 16) {
        int b = t >> 2, e = t & 3;
        float acc = 0.f;
        float sc = scale[b * 2 + sidx];
        for (int c = 0; c < 4; ++c) acc += sc * xmean[b * 4 + c] * wr[e * 516 + c];
        for (int k = 0; k < 512; ++k) acc += temb[b * 512 + k] * wr[e * 516 + 4 + k];
        acc /= zeta[0];
        if (mask[b * 4 + e] == 0) acc = -1e9f;
        lg[t] = acc;
    }
    __syncthreads();
    if (t < 4) {
        int b = t;
        float m = -1e30f;
        #pragma unroll
        for (int e = 0; e < 4; ++e) m = fmaxf(m, lg[b * 4 + e]);
        float p[4]; float s = 0.f;
        #pragma unroll
        for (int e = 0; e < 4; ++e) { p[e] = __expf(lg[b * 4 + e] - m); s += p[e]; }
        #pragma unroll
        for (int e = 0; e < 4; ++e) { p[e] /= s; probs[b * 4 + e] = p[e]; }
        int i1 = 0;
        #pragma unroll
        for (int e = 1; e < 4; ++e) if (p[e] > p[i1]) i1 = e;
        int i2 = -1;
        #pragma unroll
        for (int e = 0; e < 4; ++e) if (e != i1 && (i2 < 0 || p[e] > p[i2])) i2 = e;
        float ss = p[i1] + p[i2];
        #pragma unroll
        for (int e = 0; e < 4; ++e) rw[b * 4 + e] = 0.f;
        rw[b * 4 + i1] = p[i1] / ss;
        rw[b * 4 + i2] = p[i2] / ss;
    }
}

__global__ void textp_kernel(const float* __restrict__ te, float* __restrict__ tp) {
    int idx = blockIdx.x * 256 + threadIdx.x;
    if (idx >= 3072) return;
    int b = idx / 768, f = idx - b * 768;
    float s = 0.f;
    for (int t = 0; t < 77; ++t) s += te[((long)b * 77 + t) * 768 + f];
    tp[idx] = s * (1.f / 77.f);
}

__global__ void cond_kernel(const float* __restrict__ temb, const float* __restrict__ textp,
                            const float* __restrict__ wt, const float* __restrict__ wx,
                            float* __restrict__ cond) {
    int idx = blockIdx.x * 256 + threadIdx.x;
    if (idx >= 8192) return;
    int e = idx >> 11; int r2 = idx & 2047; int b = r2 >> 9; int n = r2 & 511;
    const float* wtp = wt + ((long)e * 512 + n) * 512;
    const float* wxp = wx + ((long)e * 512 + n) * 768;
    float a = 0.f;
    for (int k = 0; k < 512; ++k) a += temb[b * 512 + k] * wtp[k];
    for (int k = 0; k < 768; ++k) a += textp[b * 768 + k] * wxp[k];
    cond[(long)(e * 4 + b) * 512 + n] = a;
}

// UNet conv_in (4->192) + time modulation + mp_silu, f16 NHWC out (per expert)
__global__ void conv_in_kernel(const float* __restrict__ xin, const float* __restrict__ wn,
                               const float* __restrict__ tvec, _Float16* __restrict__ h1) {
    long idx = (long)blockIdx.x * 256 + threadIdx.x;
    if (idx >= 3145728) return;
    int c = (int)(idx % 192); long m = idx / 192;
    int b = (int)(m >> 12); int rem = (int)(m & 4095); int y = rem >> 6, x0 = rem & 63;
    float acc = 0.f;
    const float* wp = wn + (long)c * 36;
    for (int ci = 0; ci < 4; ++ci)
        for (int ky = 0; ky < 3; ++ky) {
            int yy = y + ky - 1; if (yy < 0 || yy > 63) continue;
            for (int kx = 0; kx < 3; ++kx) {
                int xx = x0 + kx - 1; if (xx < 0 || xx > 63) continue;
                acc += xin[((long)(b * 4 + ci) * 64 + yy) * 64 + xx] * wp[ci * 9 + ky * 3 + kx];
            }
        }
    float t = tvec[b * 192 + c];
    h1[(long)m * 192 + c] = (_Float16)mp_silu_f(acc * (1.f + t));
}

// UNet conv_out (192->4), router-weighted accumulate into out_unet [b][c][p]
__global__ void conv_out_kernel(const _Float16* __restrict__ h2, const float* __restrict__ wn,
                                const float* __restrict__ rw, int e, float* __restrict__ outp) {
    int idx = blockIdx.x * 256 + threadIdx.x;
    if (idx >= 65536) return;
    int b = idx >> 14, c = (idx >> 12) & 3, p = idx & 4095;
    int y = p >> 6, x0 = p & 63;
    float acc = 0.f;
    for (int ky = 0; ky < 3; ++ky) {
        int yy = y + ky - 1; if (yy < 0 || yy > 63) continue;
        for (int kx = 0; kx < 3; ++kx) {
            int xx = x0 + kx - 1; if (xx < 0 || xx > 63) continue;
            const _Float16* hp = h2 + (long)((b * 64 + yy) * 64 + xx) * 192;
            const float* wp = wn + (long)c * 1728 + (ky * 3 + kx);
            float s = 0.f;
            for (int ci = 0; ci < 192; ++ci) s += (float)hp[ci] * wp[ci * 9];
            acc += s;
        }
    }
    float g = rw[b * 4 + e] * acc;
    if (e == 0) outp[idx] = g; else outp[idx] += g;
}

__global__ void patchify_kernel(const float* __restrict__ xin, _Float16* __restrict__ ap) {
    int idx = blockIdx.x * 256 + threadIdx.x;
    if (idx >= 65536) return;
    int f = idx & 255; int row = idx >> 8;
    int b = row >> 6, s = row & 63;
    int gy = s >> 3, gx = s & 7;
    int c = f >> 6; int rr = f & 63; int py = rr >> 3, px = rr & 7;
    ap[idx] = (_Float16)xin[((long)(b * 4 + c) * 64 + gy * 8 + py) * 64 + gx * 8 + px];
}

// tok = (tok + pos) * (1 + cond)
__global__ void tokfix_kernel(_Float16* __restrict__ tok, const float* __restrict__ pos,
                              const float* __restrict__ cond) {
    int idx = blockIdx.x * 256 + threadIdx.x;
    if (idx >= 524288) return;
    int n = idx & 511; int row = idx >> 9; int e = row >> 8; int bs = row & 255;
    int b = bs >> 6, s = bs & 63;
    float v = (float)tok[idx] + pos[((long)e * 64 + s) * 512 + n];
    tok[idx] = (_Float16)(v * (1.f + cond[(long)(e * 4 + b) * 512 + n]));
}

__global__ void residual_kernel(_Float16* __restrict__ tok, const _Float16* __restrict__ tmp) {
    int idx = blockIdx.x * 256 + threadIdx.x;
    if (idx >= 524288) return;
    tok[idx] = (_Float16)(((float)tok[idx] + (float)tmp[idx]) * 0.70710678119f);
}

// ViT self-attention per (expert, batch, head): S=64, hd=64
__global__ __launch_bounds__(64) void vit_attn_kernel(const _Float16* __restrict__ qkv,
                                                      _Float16* __restrict__ obuf) {
    __shared__ _Float16 lq[64 * 66], lk[64 * 66], lv[64 * 66];
    __shared__ float ssc[64 * 65];
    int blk = blockIdx.x;
    int h = blk & 7, b = (blk >> 3) & 3, e = blk >> 5;
    const _Float16* base = qkv + (long)(e * 4 + b) * 64 * 1536;
    int t = threadIdx.x;
    for (int d = 0; d < 64; ++d) {
        lq[t * 66 + d] = base[(long)t * 1536 + h * 64 + d];
        lk[t * 66 + d] = base[(long)t * 1536 + 512 + h * 64 + d];
        lv[t * 66 + d] = base[(long)t * 1536 + 1024 + h * 64 + d];
    }
    __syncthreads();
    float m = -1e30f;
    for (int j = 0; j < 64; ++j) {
        float s = 0.f;
        for (int d = 0; d < 64; ++d) s += (float)lq[t * 66 + d] * (float)lk[j * 66 + d];
        s *= 0.125f;
        ssc[t * 65 + j] = s;
        m = fmaxf(m, s);
    }
    float l = 0.f;
    for (int j = 0; j < 64; ++j) { float w = __expf(ssc[t * 65 + j] - m); ssc[t * 65 + j] = w; l += w; }
    float inv = 1.f / l;
    _Float16* op = obuf + ((long)(e * 4 + b) * 64 + t) * 512 + h * 64;
    for (int d = 0; d < 64; ++d) {
        float a = 0.f;
        for (int j = 0; j < 64; ++j) a += ssc[t * 65 + j] * (float)lv[j * 66 + d];
        op[d] = (_Float16)(a * inv);
    }
}

// unpatchify + router-weighted sum -> out_vit [b][c][p]
__global__ void vit_combine_kernel(const _Float16* __restrict__ up, const float* __restrict__ rv,
                                   float* __restrict__ outv) {
    int idx = blockIdx.x * 256 + threadIdx.x;
    if (idx >= 65536) return;
    int b = idx >> 14, c = (idx >> 12) & 3, p = idx & 4095;
    int y = p >> 6, xx = p & 63;
    int gy = y >> 3, py = y & 7, gx = xx >> 3, px = xx & 7;
    int s = gy * 8 + gx; int f = (c * 8 + py) * 8 + px;
    float a = 0.f;
    #pragma unroll
    for (int e = 0; e < 4; ++e) a += rv[b * 4 + e] * (float)up[((long)(e * 4 + b) * 64 + s) * 256 + f];
    outv[idx] = a;
}

// Flash-style pixel cross-attention: 4096 queries x 4096 keys, feature dim 4
__global__ __launch_bounds__(256) void cross_attn_kernel(
    const float* __restrict__ out_unet, const float* __restrict__ out_vit,
    const float* __restrict__ wq, const float* __restrict__ wk,
    const float* __restrict__ wv, const float* __restrict__ wo,
    float* __restrict__ attn_img) {
    __shared__ float lk[256 * 4], lv[256 * 4];
    int b = blockIdx.y;
    int p = blockIdx.x * 256 + threadIdx.x;
    const float* U = out_unet + (long)b * 16384;
    const float* V = out_vit + (long)b * 16384;
    float qt[4], qv[4];
    #pragma unroll
    for (int c = 0; c < 4; ++c) qt[c] = U[c * 4096 + p];
    #pragma unroll
    for (int i = 0; i < 4; ++i)
        qv[i] = qt[0] * wq[i * 4 + 0] + qt[1] * wq[i * 4 + 1] + qt[2] * wq[i * 4 + 2] + qt[3] * wq[i * 4 + 3];
    float m = -1e30f, l = 0.f, o0 = 0.f, o1 = 0.f, o2 = 0.f, o3 = 0.f;
    for (int kt = 0; kt < 4096; kt += 256) {
        int pk = kt + threadIdx.x;
        float ct[4];
        #pragma unroll
        for (int c = 0; c < 4; ++c) ct[c] = V[c * 4096 + pk];
        #pragma unroll
        for (int i = 0; i < 4; ++i) {
            lk[threadIdx.x * 4 + i] = ct[0] * wk[i * 4 + 0] + ct[1] * wk[i * 4 + 1] + ct[2] * wk[i * 4 + 2] + ct[3] * wk[i * 4 + 3];
            lv[threadIdx.x * 4 + i] = ct[0] * wv[i * 4 + 0] + ct[1] * wv[i * 4 + 1] + ct[2] * wv[i * 4 + 2] + ct[3] * wv[i * 4 + 3];
        }
        __syncthreads();
        for (int j = 0; j < 256; ++j) {
            float s = 0.5f * (qv[0] * lk[j * 4 + 0] + qv[1] * lk[j * 4 + 1] + qv[2] * lk[j * 4 + 2] + qv[3] * lk[j * 4 + 3]);
            float mn = fmaxf(m, s);
            float corr = __expf(m - mn), w = __expf(s - mn);
            l = l * corr + w;
            o0 = o0 * corr + w * lv[j * 4 + 0];
            o1 = o1 * corr + w * lv[j * 4 + 1];
            o2 = o2 * corr + w * lv[j * 4 + 2];
            o3 = o3 * corr + w * lv[j * 4 + 3];
            m = mn;
        }
        __syncthreads();
    }
    float inv = 1.f / l;
    o0 *= inv; o1 *= inv; o2 *= inv; o3 *= inv;
    float op[4];
    #pragma unroll
    for (int i = 0; i < 4; ++i)
        op[i] = o0 * wo[i * 4 + 0] + o1 * wo[i * 4 + 1] + o2 * wo[i * 4 + 2] + o3 * wo[i * 4 + 3];
    const float t = 0.8f;
    float dn = rsqrtf((1.f - t) * (1.f - t) + t * t);
    #pragma unroll
    for (int c = 0; c < 4; ++c)
        attn_img[(long)b * 16384 + c * 4096 + p] = ((1.f - t) * qt[c] + t * op[c]) * dn;
}

// gated combine: 1x1 conv 8->4, mp_silu, 1x1 conv 4->2, softmax over channels, mix
__global__ void gate_kernel(const float* __restrict__ out_unet, const float* __restrict__ attn_img,
                            const float* __restrict__ g1w, const float* __restrict__ g2w,
                            float* __restrict__ out) {
    int idx = blockIdx.x * 256 + threadIdx.x;
    if (idx >= 16384) return;
    int b = idx >> 12, p = idx & 4095;
    float u[4], a[4];
    #pragma unroll
    for (int c = 0; c < 4; ++c) {
        u[c] = out_unet[(long)(b * 4 + c) * 4096 + p];
        a[c] = attn_img[(long)(b * 4 + c) * 4096 + p];
    }
    float s1[4];
    #pragma unroll
    for (int k = 0; k < 4; ++k) {
        float g = 0.f;
        #pragma unroll
        for (int c = 0; c < 4; ++c) g += g1w[k * 8 + c] * u[c] + g1w[k * 8 + 4 + c] * a[c];
        s1[k] = mp_silu_f(g);
    }
    float g2[2];
    #pragma unroll
    for (int j = 0; j < 2; ++j) {
        float g = 0.f;
        #pragma unroll
        for (int k = 0; k < 4; ++k) g += g2w[j * 4 + k] * s1[k];
        g2[j] = g;
    }
    float mm = fmaxf(g2[0], g2[1]);
    float e0 = __expf(g2[0] - mm), e1 = __expf(g2[1] - mm);
    float si = 1.f / (e0 + e1);
    e0 *= si; e1 *= si;
    #pragma unroll
    for (int c = 0; c < 4; ++c)
        out[(long)(b * 4 + c) * 4096 + p] = e0 * u[c] + e1 * a[c];
}

// ---------------------------------------------------------------------------
extern "C" void kernel_launch(void* const* d_in, const int* in_sizes, int n_in,
                              void* d_out, int out_size, void* d_ws, size_t ws_size,
                              hipStream_t stream) {
    (void)in_sizes; (void)n_in; (void)out_size; (void)ws_size;
    const float* x        = (const float*)d_in[0];
    const float* time_vec = (const float*)d_in[1];
    const float* text_emb = (const float*)d_in[2];
    const unsigned char* umask = (const unsigned char*)d_in[3];
    const unsigned char* vmask = (const unsigned char*)d_in[4];
    const float* zeta     = (const float*)d_in[5];
    const float* ffreq    = (const float*)d_in[6];
    const float* fphase   = (const float*)d_in[7];
    const float* w_f1     = (const float*)d_in[8];
    const float* w_f2     = (const float*)d_in[9];
    const float* w_scale  = (const float*)d_in[10];
    const float* w_ru     = (const float*)d_in[11];
    const float* w_rv     = (const float*)d_in[12];
    const float* uw_in    = (const float*)d_in[13];
    const float* uw_time  = (const float*)d_in[14];
    const float* uw_mid   = (const float*)d_in[15];
    const float* uw_out   = (const float*)d_in[16];
    const float* vw_patch = (const float*)d_in[17];
    const float* v_pos    = (const float*)d_in[18];
    const float* vw_qkv   = (const float*)d_in[19];
    const float* vw_proj  = (const float*)d_in[20];
    const float* vw_time  = (const float*)d_in[21];
    const float* vw_text  = (const float*)d_in[22];
    const float* vw_mlp1  = (const float*)d_in[23];
    const float* vw_mlp2  = (const float*)d_in[24];
    const float* vw_unp   = (const float*)d_in[25];
    const float* ca_wq    = (const float*)d_in[26];
    const float* ca_wk    = (const float*)d_in[27];
    const float* ca_wv    = (const float*)d_in[28];
    const float* ca_wo    = (const float*)d_in[29];
    const float* gw1      = (const float*)d_in[30];
    const float* gw2      = (const float*)d_in[31];
    float* out = (float*)d_out;

    // workspace carve
    char* base = (char*)d_ws; size_t off = 0;
    auto alloc = [&](size_t bytes) -> void* {
        void* p = base + off;
        off = (off + bytes + 255) & ~(size_t)255;
        return p;
    };
    // normalized f32 weights
    float* nw_f1    = (float*)alloc(262144 * 4);
    float* nw_f2    = (float*)alloc(524288 * 4);
    float* nw_scale = (float*)alloc(1024 * 4);
    float* nw_ru    = (float*)alloc(2064 * 4);
    float* nw_rv    = (float*)alloc(2064 * 4);
    float* nuw_in   = (float*)alloc(27648 * 4);
    float* nuw_time = (float*)alloc(393216 * 4);
    float* nuw_out  = (float*)alloc(27648 * 4);
    float* nvw_time = (float*)alloc(1048576 * 4);
    float* nvw_text = (float*)alloc(1572864 * 4);
    float* ncaq = (float*)alloc(16 * 4);
    float* ncak = (float*)alloc(16 * 4);
    float* ncav = (float*)alloc(16 * 4);
    float* ncao = (float*)alloc(16 * 4);
    float* ngw1 = (float*)alloc(32 * 4);
    float* ngw2 = (float*)alloc(8 * 4);
    // normalized f16 weights
    _Float16* wmid      = (_Float16*)alloc((size_t)1327104 * 2);
    _Float16* nvw_patch = (_Float16*)alloc((size_t)524288 * 2);
    _Float16* nvw_qkv   = (_Float16*)alloc((size_t)3145728 * 2);
    _Float16* nvw_proj  = (_Float16*)alloc((size_t)1048576 * 2);
    _Float16* nvw_mlp1  = (_Float16*)alloc((size_t)4194304 * 2);
    _Float16* nvw_mlp2  = (_Float16*)alloc((size_t)4194304 * 2);
    _Float16* nvw_unp   = (_Float16*)alloc((size_t)524288 * 2);
    // f32 activations
    float* femb_b   = (float*)alloc(1024 * 4);
    float* h1t_b    = (float*)alloc(4096 * 4);
    float* temb_b   = (float*)alloc(2048 * 4);
    float* scale_b  = (float*)alloc(8 * 4);
    float* xmean_b  = (float*)alloc(16 * 4);
    float* textp_b  = (float*)alloc(3072 * 4);
    float* in_vit_b = (float*)alloc(65536 * 4);
    float* in_unet_b= (float*)alloc(65536 * 4);
    float* ut_b     = (float*)alloc(3072 * 4);
    float* cond_b   = (float*)alloc(8192 * 4);
    float* r_u_b    = (float*)alloc(16 * 4);
    float* r_v_b    = (float*)alloc(16 * 4);
    float* out_unet_b = (float*)alloc(65536 * 4);
    float* out_vit_b  = (float*)alloc(65536 * 4);
    float* attn_b     = (float*)alloc(65536 * 4);
    // f16 activations
    _Float16* h1_b     = (_Float16*)alloc((size_t)3145728 * 2);
    _Float16* h2_b     = (_Float16*)alloc((size_t)3145728 * 2);
    _Float16* apatch_b = (_Float16*)alloc((size_t)65536 * 2);
    _Float16* tok_b    = (_Float16*)alloc((size_t)524288 * 2);
    _Float16* qkv_b    = (_Float16*)alloc((size_t)1572864 * 2);
    _Float16* obuf_b   = (_Float16*)alloc((size_t)524288 * 2);
    _Float16* tmp512_b = (_Float16*)alloc((size_t)524288 * 2);
    _Float16* mlph_b   = (_Float16*)alloc((size_t)2097152 * 2);
    _Float16* up_b     = (_Float16*)alloc((size_t)262144 * 2);

    // ---- weight normalization ----
    norm_rows_f32<<<1024, 256, 0, stream>>>(w_f1, nw_f1, 1024, 256);
    norm_rows_f32<<<512, 256, 0, stream>>>(w_f2, nw_f2, 512, 1024);
    norm_rows_f32<<<2, 256, 0, stream>>>(w_scale, nw_scale, 2, 512);
    norm_rows_f32<<<4, 256, 0, stream>>>(w_ru, nw_ru, 4, 516);
    norm_rows_f32<<<4, 256, 0, stream>>>(w_rv, nw_rv, 4, 516);
    norm_rows_f32<<<768, 256, 0, stream>>>(uw_in, nuw_in, 768, 36);
    norm_rows_f32<<<768, 256, 0, stream>>>(uw_time, nuw_time, 768, 512);
    norm_repack_uwmid<<<768, 256, 0, stream>>>(uw_mid, wmid);
    norm_rows_f32<<<16, 256, 0, stream>>>(uw_out, nuw_out, 16, 1728);
    norm_rows_f16<<<2048, 256, 0, stream>>>(vw_patch, nvw_patch, 2048, 256);
    norm_rows_f16<<<6144, 256, 0, stream>>>(vw_qkv, nvw_qkv, 6144, 512);
    norm_rows_f16<<<2048, 256, 0, stream>>>(vw_proj, nvw_proj, 2048, 512);
    norm_rows_f32<<<2048, 256, 0, stream>>>(vw_time, nvw_time, 2048, 512);
    norm_rows_f32<<<2048, 256, 0, stream>>>(vw_text, nvw_text, 2048, 768);
    norm_rows_f16<<<8192, 256, 0, stream>>>(vw_mlp1, nvw_mlp1, 8192, 512);
    norm_rows_f16<<<2048, 256, 0, stream>>>(vw_mlp2, nvw_mlp2, 2048, 2048);
    norm_rows_f16<<<1024, 256, 0, stream>>>(vw_unp, nvw_unp, 1024, 512);
    norm_rows_f32<<<4, 256, 0, stream>>>(ca_wq, ncaq, 4, 4);
    norm_rows_f32<<<4, 256, 0, stream>>>(ca_wk, ncak, 4, 4);
    norm_rows_f32<<<4, 256, 0, stream>>>(ca_wv, ncav, 4, 4);
    norm_rows_f32<<<4, 256, 0, stream>>>(ca_wo, ncao, 4, 4);
    norm_rows_f32<<<4, 256, 0, stream>>>(gw1, ngw1, 4, 8);
    norm_rows_f32<<<2, 256, 0, stream>>>(gw2, ngw2, 2, 4);

    // ---- time embedding, scaling router, routers ----
    femb_kernel<<<4, 256, 0, stream>>>(time_vec, ffreq, fphase, femb_b);
    small_linear_kernel<<<16, 256, 0, stream>>>(femb_b, nw_f1, h1t_b, 4, 1024, 256, 1);
    small_linear_kernel<<<8, 256, 0, stream>>>(h1t_b, nw_f2, temb_b, 4, 512, 1024, 0);
    scale_kernel<<<1, 32, 0, stream>>>(temb_b, nw_scale, zeta, scale_b);
    xmean_kernel<<<16, 256, 0, stream>>>(x, xmean_b);
    inscale_kernel<<<256, 256, 0, stream>>>(x, scale_b, in_vit_b, in_unet_b);
    router_kernel<<<1, 64, 0, stream>>>(xmean_b, scale_b, 1, temb_b, nw_ru, umask, zeta, r_u_b, out + 65536);
    router_kernel<<<1, 64, 0, stream>>>(xmean_b, scale_b, 0, temb_b, nw_rv, vmask, zeta, r_v_b, out + 65552);
    textp_kernel<<<12, 256, 0, stream>>>(text_emb, textp_b);
    for (int e = 0; e < 4; ++e)
        small_linear_kernel<<<3, 256, 0, stream>>>(temb_b, nuw_time + (long)e * 192 * 512,
                                                   ut_b + e * 768, 4, 192, 512, 0);
    cond_kernel<<<32, 256, 0, stream>>>(temb_b, textp_b, nvw_time, nvw_text, cond_b);

    // ---- UNet experts (per-expert loop reusing h1/h2) ----
    for (int e = 0; e < 4; ++e) {
        conv_in_kernel<<<12288, 256, 0, stream>>>(in_unet_b, nuw_in + (long)e * 192 * 36,
                                                  ut_b + e * 768, h1_b);
        conv_mid_wmma_kernel<<<dim3(256, 3), 128, 0, stream>>>(h1_b, wmid + (long)e * 192 * 1728, h2_b);
        conv_out_kernel<<<256, 256, 0, stream>>>(h2_b, nuw_out + (long)e * 4 * 1728, r_u_b, e, out_unet_b);
    }

    // ---- ViT experts (batched over experts via grid.z) ----
    patchify_kernel<<<256, 256, 0, stream>>>(in_vit_b, apatch_b);
    gemm_wmma_kernel<0><<<dim3(4, 8, 4), 128, 0, stream>>>(apatch_b, (long)0, nvw_patch, (long)512 * 256,
                                                           tok_b, (long)256 * 512, 256, 512, 256);
    tokfix_kernel<<<2048, 256, 0, stream>>>(tok_b, v_pos, cond_b);
    gemm_wmma_kernel<0><<<dim3(4, 24, 4), 128, 0, stream>>>(tok_b, (long)256 * 512, nvw_qkv, (long)1536 * 512,
                                                            qkv_b, (long)256 * 1536, 256, 1536, 512);
    vit_attn_kernel<<<128, 64, 0, stream>>>(qkv_b, obuf_b);
    gemm_wmma_kernel<0><<<dim3(4, 8, 4), 128, 0, stream>>>(obuf_b, (long)256 * 512, nvw_proj, (long)512 * 512,
                                                           tmp512_b, (long)256 * 512, 256, 512, 512);
    residual_kernel<<<2048, 256, 0, stream>>>(tok_b, tmp512_b);
    gemm_wmma_kernel<1><<<dim3(4, 32, 4), 128, 0, stream>>>(tok_b, (long)256 * 512, nvw_mlp1, (long)2048 * 512,
                                                            mlph_b, (long)256 * 2048, 256, 2048, 512);
    gemm_wmma_kernel<0><<<dim3(4, 8, 4), 128, 0, stream>>>(mlph_b, (long)256 * 2048, nvw_mlp2, (long)512 * 2048,
                                                           tmp512_b, (long)256 * 512, 256, 512, 2048);
    residual_kernel<<<2048, 256, 0, stream>>>(tok_b, tmp512_b);
    gemm_wmma_kernel<0><<<dim3(4, 4, 4), 128, 0, stream>>>(tok_b, (long)256 * 512, nvw_unp, (long)256 * 512,
                                                           up_b, (long)256 * 256, 256, 256, 512);
    vit_combine_kernel<<<256, 256, 0, stream>>>(up_b, r_v_b, out_vit_b);

    // ---- pixel cross-attention + gated combine ----
    cross_attn_kernel<<<dim3(16, 4), 256, 0, stream>>>(out_unet_b, out_vit_b, ncaq, ncak, ncav, ncao, attn_b);
    gate_kernel<<<64, 256, 0, stream>>>(out_unet_b, attn_b, ngw1, ngw2, out);
}
